// InitialNetwork_3702261809580
// MI455X (gfx1250) — compile-verified, matrix operand order checked by reading, not run on hardware
//
#include <hip/hip_runtime.h>
#include <hip/hip_bf16.h>

typedef __attribute__((ext_vector_type(16))) _Float16 v16h;
typedef __attribute__((ext_vector_type(8)))  float    v8f;

#define BT 16          // batch tile (WMMA M)
#define TB 512         // time steps
#define DD 128         // input dim
#define HH 128         // hidden
#define GG 512         // 4*H

__device__ __forceinline__ float sigf(float x) { return 1.0f / (1.0f + __expf(-x)); }
// tanh(x) = 1 - 2/(exp(2x)+1): hardware v_exp_f32 path, saturates to +/-1.
__device__ __forceinline__ float tanh_fast(float x) {
    float e = __expf(2.0f * x);
    return 1.0f - 2.0f / (e + 1.0f);
}

// ---------------------------------------------------------------------------
// Prep: xs f32 -> f16 (so WMMA A-fragments of x_t are direct 32B global loads)
// ---------------------------------------------------------------------------
__global__ void cvt_f32_f16(const float* __restrict__ src, _Float16* __restrict__ dst, int n) {
    for (int i = blockIdx.x * blockDim.x + threadIdx.x; i < n; i += gridDim.x * blockDim.x)
        dst[i] = (_Float16)src[i];
}

// ---------------------------------------------------------------------------
// Prep: dst[n*K + k] = (f16) (k < Ka ? A[k*N+n] : B[(k-Ka)*N+n])
// B-matrix-friendly [N][K] f16 layout (lane = column, contiguous K).
// ---------------------------------------------------------------------------
__global__ void transpose_concat_f16(const float* __restrict__ A, int Ka,
                                     const float* __restrict__ B, int Kb,
                                     int N, _Float16* __restrict__ dst) {
    int K = Ka + Kb;
    int total = N * K;
    for (int idx = blockIdx.x * blockDim.x + threadIdx.x; idx < total;
         idx += gridDim.x * blockDim.x) {
        int n = idx / K;
        int k = idx - n * K;
        float v = (k < Ka) ? A[k * N + n] : B[(k - Ka) * N + n];
        dst[idx] = (_Float16)v;
    }
}

// ---------------------------------------------------------------------------
// Forward LSTM scan, gate-aligned + software-pipelined:
// 16 WGs x 256 threads (8 waves). Wave w owns hidden slice j in [16w,16w+16)
// for ALL four gates -> cell nonlinearity entirely in registers.
//  - Wh^T fragments: 128 VGPRs, loaded once (loop invariant)
//  - Wx^T: 128KB LDS-resident
//  - x-GEMM for step t+1 computed during step t (off the h-recurrence chain;
//    XDL WMMAs co-execute with the gate transcendental VALU)
//  - h: 2x4KB double-buffered LDS -> ONE barrier per step
// ---------------------------------------------------------------------------
__global__ __launch_bounds__(256) void lstm_fwd_scan(
    const _Float16* __restrict__ xs16, const float* __restrict__ bvec,
    const _Float16* __restrict__ wtx,  // Wx^T [512][128] f16
    const _Float16* __restrict__ wth,  // Wh^T [512][128] f16
    float* __restrict__ hf_out) {
    __shared__ __align__(32) _Float16 Wxsm[GG * DD];     // 128 KB [N=512][K=128]
    __shared__ __align__(32) _Float16 Hbuf[2][BT * HH];  // 2 x 4 KB

    const int tid = threadIdx.x;
    const int b0  = blockIdx.x * BT;
    const int w = tid >> 5, lane = tid & 31;
    const int mA = lane & 15;
    const int khalf = (lane >> 4) * 16;
    const int mbase = (lane >> 4) * 8;

    // Stage Wx^T into LDS (once).
    {
        const uint4* src = (const uint4*)wtx;
        uint4* dst = (uint4*)Wxsm;
        for (int i = tid; i < (GG * DD * 2) / 16; i += 256) dst[i] = src[i];
    }
    // h0 = 0
    for (int i = tid; i < BT * HH; i += 256) Hbuf[0][i] = (_Float16)0.0f;

    // Wh^T B-fragments -> registers (loop invariant): gate g, k-step ks
    v16h bh[4][4];
#pragma unroll
    for (int g = 0; g < 4; ++g)
#pragma unroll
        for (int ks = 0; ks < 4; ++ks) {
            const int ncol = g * HH + w * 16 + mA;
            bh[g][ks] = *(const v16h*)&wth[ncol * DD + ks * 32 + khalf];
        }

    // this lane's hidden column and biases (same j for all 8 m-rows it holds)
    const int j = w * 16 + mA;
    const float bi = bvec[j], bf = bvec[HH + j], bg = bvec[2 * HH + j], bo = bvec[3 * HH + j];
    float cst[8];
#pragma unroll
    for (int r = 0; r < 8; ++r) cst[r] = 0.0f;

    const _Float16* xrow = xs16 + (b0 + mA) * (TB * DD) + khalf;  // this lane's A rows

    __syncthreads();

    // x-GEMM for a given t into acc (fresh accumulators)
    auto xgemm = [&](v8f (&acc)[4], int t) {
#pragma unroll
        for (int g = 0; g < 4; ++g) acc[g] = (v8f){};
#pragma unroll
        for (int ks = 0; ks < 4; ++ks) {
            v16h af = *(const v16h*)&xrow[t * DD + ks * 32];
#pragma unroll
            for (int g = 0; g < 4; ++g) {
                v16h bx = *(const v16h*)&Wxsm[(g * HH + j) * DD + ks * 32 + khalf];
                acc[g] = __builtin_amdgcn_wmma_f32_16x16x32_f16(
                    false, af, false, bx, (short)0, acc[g], false, false);
            }
        }
    };

    // one recurrence step: acc holds x-part of step t; accn gets x-part of t+1
    auto step = [&](v8f (&acc)[4], v8f (&accn)[4], int p, int t) {
        // h-part (the true serial dependency): A from Hbuf[p], B from registers
#pragma unroll
        for (int ks = 0; ks < 4; ++ks) {
            v16h af = *(const v16h*)&Hbuf[p][mA * HH + ks * 32 + khalf];
#pragma unroll
            for (int g = 0; g < 4; ++g)
                acc[g] = __builtin_amdgcn_wmma_f32_16x16x32_f16(
                    false, af, false, bh[g][ks], (short)0, acc[g], false, false);
        }
        // prefetch-compute x-part of t+1 (independent; co-executes with gates).
        // At t=TB-1 this reads up to 256 B past xs16 (still inside d_ws) and is discarded.
        xgemm(accn, t + 1);
        // cell update fully in registers: lane holds (m = mbase+r, j)
#pragma unroll
        for (int r = 0; r < 8; ++r) {
            float zi = acc[0][r] + bi;
            float zf = acc[1][r] + bf;
            float zg = acc[2][r] + bg;
            float zo = acc[3][r] + bo;
            float c = sigf(zf) * cst[r] + sigf(zi) * tanh_fast(zg);
            cst[r] = c;
            float h = sigf(zo) * tanh_fast(c);
            Hbuf[p ^ 1][(mbase + r) * HH + j] = (_Float16)h;
            if (t == TB - 1) hf_out[(b0 + mbase + r) * HH + j] = h;
        }
        __syncthreads();   // single barrier per step (double-buffered h)
    };

    v8f accA[4], accB[4];
    xgemm(accA, 0);
    for (int t = 0; t < TB; t += 2) {        // 2x unroll: ping-pong acc sets
        step(accA, accB, 0, t);
        step(accB, accA, 1, t + 1);
    }
}

// ---------------------------------------------------------------------------
// Backward direction collapses to ONE cell step on x_{T-1} (zero carry):
// z = x_{T-1} @ Wx_b + b;  c = sig(i)*tanh(g);  h = sig(o)*tanh(c)
// ---------------------------------------------------------------------------
__global__ __launch_bounds__(256) void lstm_bwd_laststep(
    const float* __restrict__ xs, const float* __restrict__ bvec,
    const _Float16* __restrict__ wt, float* __restrict__ hb_out) {
    __shared__ __align__(32) _Float16 Wsm[GG * DD];   // 128 KB
    __shared__ __align__(32) _Float16 Asm[BT * DD];   //   4 KB
    __shared__ __align__(16) float    Zsm[BT * GG];   //  32 KB

    const int tid = threadIdx.x;
    const int b0  = blockIdx.x * BT;
    const int j = tid & 127, bpar = tid >> 7;
    const int w = tid >> 5, lane = tid & 31;
    const int mA = lane & 15, khalf = (lane >> 4) * 16, mbase = (lane >> 4) * 8;

    {
        const uint4* src = (const uint4*)wt;
        uint4* dst = (uint4*)Wsm;
        for (int i = tid; i < (GG * DD * 2) / 16; i += 256) dst[i] = src[i];
    }
#pragma unroll
    for (int q = 0; q < 8; ++q) {
        int brow = bpar + 2 * q;
        Asm[brow * DD + j] = (_Float16)xs[((b0 + brow) * TB + (TB - 1)) * DD + j];
    }
    __syncthreads();

    v8f acc[4] = {{}, {}, {}, {}};
#pragma unroll
    for (int ks = 0; ks < 4; ++ks) {           // K = 128
        const int k0 = ks * 32;
        v16h afrag = *(const v16h*)&Asm[mA * DD + k0 + khalf];
#pragma unroll
        for (int nt = 0; nt < 4; ++nt) {
            const int ncol = w * 64 + nt * 16 + mA;
            v16h bfrag = *(const v16h*)&Wsm[ncol * DD + k0 + khalf];
            acc[nt] = __builtin_amdgcn_wmma_f32_16x16x32_f16(
                false, afrag, false, bfrag, (short)0, acc[nt], false, false);
        }
    }
#pragma unroll
    for (int nt = 0; nt < 4; ++nt)
#pragma unroll
        for (int r = 0; r < 8; ++r)
            Zsm[(mbase + r) * GG + w * 64 + nt * 16 + mA] = acc[nt][r];
    __syncthreads();

#pragma unroll
    for (int q = 0; q < 8; ++q) {
        int brow = bpar + 2 * q;
        float zi = Zsm[brow * GG + j] + bvec[j];
        float zg = Zsm[brow * GG + 2 * HH + j] + bvec[2 * HH + j];
        float zo = Zsm[brow * GG + 3 * HH + j] + bvec[3 * HH + j];
        float c = sigf(zi) * tanh_fast(zg);           // f-gate * c0 == 0
        hb_out[(b0 + brow) * HH + j] = sigf(zo) * tanh_fast(c);
    }
}

// ---------------------------------------------------------------------------
// Head: out = relu([hf,hb] @ W1 + b1) @ W2 + b2 ; split mean/logvar.
// ---------------------------------------------------------------------------
__global__ __launch_bounds__(256) void head_kernel(
    const float* __restrict__ hf, const float* __restrict__ hb,
    const float* __restrict__ b1, const float* __restrict__ b2,
    const _Float16* __restrict__ w1t, const _Float16* __restrict__ w2t,
    float* __restrict__ outp) {
    __shared__ __align__(32) _Float16 W1sm[128 * 256]; // 64 KB [N=128][K=256]
    __shared__ __align__(32) _Float16 W2sm[64 * 128];  // 16 KB [N=64][K=128]
    __shared__ __align__(32) _Float16 Asm[BT * 256];
    __shared__ __align__(32) _Float16 A2sm[BT * 128];

    const int tid = threadIdx.x;
    const int b0  = blockIdx.x * BT;
    const int w = tid >> 5, lane = tid & 31;
    const int mA = lane & 15, khalf = (lane >> 4) * 16, mbase = (lane >> 4) * 8;

    {
        const uint4* s1 = (const uint4*)w1t; uint4* d1 = (uint4*)W1sm;
        for (int i = tid; i < 4096; i += 256) d1[i] = s1[i];
        const uint4* s2 = (const uint4*)w2t; uint4* d2 = (uint4*)W2sm;
        for (int i = tid; i < 1024; i += 256) d2[i] = s2[i];
    }
#pragma unroll
    for (int q = 0; q < BT; ++q)   // A[q][tid] = concat(hf, hb)
        Asm[q * 256 + tid] = (_Float16)((tid < 128) ? hf[(b0 + q) * 128 + tid]
                                                    : hb[(b0 + q) * 128 + tid - 128]);
    __syncthreads();

    // GEMM1: wave w -> cols [16w, 16w+16), K=256
    {
        v8f acc = {};
#pragma unroll
        for (int ks = 0; ks < 8; ++ks) {
            const int k0 = ks * 32;
            v16h afrag = *(const v16h*)&Asm[mA * 256 + k0 + khalf];
            v16h bfrag = *(const v16h*)&W1sm[(w * 16 + mA) * 256 + k0 + khalf];
            acc = __builtin_amdgcn_wmma_f32_16x16x32_f16(
                false, afrag, false, bfrag, (short)0, acc, false, false);
        }
        const int col = w * 16 + mA;
        const float bias = b1[col];
#pragma unroll
        for (int r = 0; r < 8; ++r) {
            float v = acc[r] + bias;
            v = v > 0.0f ? v : 0.0f;
            A2sm[(mbase + r) * 128 + col] = (_Float16)v;
        }
    }
    __syncthreads();

    // GEMM2: waves 0..3 -> 64 output cols, K=128; mean/logvar split
    if (w < 4) {
        v8f acc = {};
#pragma unroll
        for (int ks = 0; ks < 4; ++ks) {
            const int k0 = ks * 32;
            v16h afrag = *(const v16h*)&A2sm[mA * 128 + k0 + khalf];
            v16h bfrag = *(const v16h*)&W2sm[(w * 16 + mA) * 128 + k0 + khalf];
            acc = __builtin_amdgcn_wmma_f32_16x16x32_f16(
                false, afrag, false, bfrag, (short)0, acc, false, false);
        }
        const int col = w * 16 + mA;
        const float bias = b2[col];
#pragma unroll
        for (int r = 0; r < 8; ++r) {
            float v = acc[r] + bias;
            int b = b0 + mbase + r;
            if (col < 32) outp[b * 32 + col] = v;                 // mean
            else          outp[8192 + b * 32 + (col - 32)] = v;   // logvar
        }
    }
}

// ---------------------------------------------------------------------------
extern "C" void kernel_launch(void* const* d_in, const int* in_sizes, int n_in,
                              void* d_out, int out_size, void* d_ws, size_t ws_size,
                              hipStream_t stream) {
    (void)in_sizes; (void)n_in; (void)out_size; (void)ws_size;
    const float* xs   = (const float*)d_in[0];
    const float* Wx_f = (const float*)d_in[1];
    const float* Wh_f = (const float*)d_in[2];
    const float* b_f  = (const float*)d_in[3];
    const float* Wx_b = (const float*)d_in[4];
    // Wh_b (d_in[5]) is dead: backward scan's last output uses zero h.
    const float* b_b  = (const float*)d_in[6];
    const float* W1   = (const float*)d_in[7];
    const float* b1   = (const float*)d_in[8];
    const float* W2   = (const float*)d_in[9];
    const float* b2   = (const float*)d_in[10];
    float* outp = (float*)d_out;

    char* ws = (char*)d_ws;
    _Float16* xs16 = (_Float16*)(ws + 0);          // [256*512*128] f16  33554432 B
    _Float16* wtxf = (_Float16*)(ws + 33554432);   // Wx_f^T [512][128]    131072 B
    _Float16* wthf = (_Float16*)(ws + 33685504);   // Wh_f^T [512][128]    131072 B
    _Float16* wtb  = (_Float16*)(ws + 33816576);   // Wx_b^T [512][128]    131072 B
    _Float16* w1t  = (_Float16*)(ws + 33947648);   // W1^T   [128][256]     65536 B
    _Float16* w2t  = (_Float16*)(ws + 34013184);   // W2^T   [64][128]      16384 B
    float*    hf   = (float*)   (ws + 34029568);   // [256][128] f32       131072 B
    float*    hb   = (float*)   (ws + 34160640);   // [256][128] f32       131072 B

    const int NXS = 256 * 512 * 128;
    cvt_f32_f16<<<8192, 256, 0, stream>>>(xs, xs16, NXS);
    transpose_concat_f16<<<(512 * 128 + 255) / 256, 256, 0, stream>>>(Wx_f, 128, Wx_f, 0, 512, wtxf);
    transpose_concat_f16<<<(512 * 128 + 255) / 256, 256, 0, stream>>>(Wh_f, 128, Wh_f, 0, 512, wthf);
    transpose_concat_f16<<<(512 * 128 + 255) / 256, 256, 0, stream>>>(Wx_b, 128, Wx_b, 0, 512, wtb);
    transpose_concat_f16<<<(128 * 256 + 255) / 256, 256, 0, stream>>>(W1, 256, W1, 0, 128, w1t);
    transpose_concat_f16<<<(64 * 128 + 255) / 256, 256, 0, stream>>>(W2, 128, W2, 0, 64, w2t);

    lstm_bwd_laststep<<<16, 256, 0, stream>>>(xs, b_b, wtb, hb);
    lstm_fwd_scan   <<<16, 256, 0, stream>>>(xs16, b_f, wtxf, wthf, hf);
    head_kernel     <<<16, 256, 0, stream>>>(hf, hb, b1, b2, w1t, w2t, outp);
}
